// DeltaSynapse_80934363725924
// MI455X (gfx1250) — matrix operand order repaired and checked by reading, not compiled, not run on hardware
//
#include <hip/hip_runtime.h>

// ---- shapes (fixed by the reference) ----
#define D_ 16
#define B_ 8
#define E_ 2048
#define O_ 2048
#define KTOT (D_ * E_)          // 32768 = flattened K dimension (d,e)
#define NCHUNK (KTOT / 32)      // 1024 K-chunks of 32
#define NTILES (O_ / 16)        // 128 N tiles
#define KSLICES 32              // K split across waves
#define CH_PER_SLICE (NCHUNK / KSLICES) // 32 chunks per wave

typedef __attribute__((ext_vector_type(16))) __bf16          v16bf;
typedef __attribute__((ext_vector_type(8)))  float           v8f;
typedef __attribute__((ext_vector_type(16))) unsigned short  u16x16;

union BF16Frag { u16x16 u; v16bf v; };

__device__ __forceinline__ unsigned short f2bf(float f) {
    unsigned int u = __builtin_bit_cast(unsigned int, f);
    u += 0x7fffu + ((u >> 16) & 1u);   // round-to-nearest-even
    return (unsigned short)(u >> 16);
}

// ---- 1) Weff = signs * W, stored as bf16 ----
__global__ void weff_kernel(const float* __restrict__ W,
                            const float* __restrict__ signs,
                            unsigned short* __restrict__ weff) {
    int i = blockIdx.x * blockDim.x + threadIdx.x;
    if (i < E_ * O_) weff[i] = f2bf(W[i] * signs[i]);
}

// ---- 2) pack S[d,b,e] = Xd*(1+Wshort) into WMMA A-fragment layout ----
// A (16-bit, 16x32), contiguous k order chosen here: lane = 16*h + M; element j <-> K = 16*h + j
// apack layout: [chunk][lane][j] bf16, chunk stride 512, lane stride 16
__global__ void apack_kernel(const float* __restrict__ Xd,
                             const float* __restrict__ Wshort,
                             unsigned short* __restrict__ apack) {
    int k = blockIdx.x * blockDim.x + threadIdx.x;
    if (k >= KTOT) return;
    int d = k >> 11, e = k & (E_ - 1);
    int chunk = k >> 5, kk = k & 31;
    int j = kk & 15;
    int h = kk >> 4;
    unsigned short* base = apack + (size_t)chunk * 512 + h * 256 + j;
    #pragma unroll
    for (int m = 0; m < 16; ++m) {
        float v = 0.0f;
        if (m < B_) {
            size_t idx = ((size_t)(d * B_ + m) << 11) + e;
            v = Xd[idx] * (Wshort[idx] + 1.0f);
        }
        base[m * 16] = f2bf(v);   // lane = 16h + m  ->  offset 256h + 16m + j
    }
}

// ---- 3) GEMM: I_partial[slice] += A(16xK) x (Weff o delaymap)(Kx16) ----
// B (16-bit, 32x16), same contiguous k order as A, chosen here: lanes 0-15 hold K=0..15 (N=lane), lanes 16-31 hold K=16..31
__global__ __launch_bounds__(256) void gemm_kernel(
        const unsigned short* __restrict__ weff,   // E*O bf16 (L2-resident)
        const unsigned short* __restrict__ apack,  // NCHUNK*512 bf16
        const float*          __restrict__ dm,     // D*E*O f32 (streamed NT)
        float*                __restrict__ partial) // KSLICES * 8 * O_
{
    const int lane   = threadIdx.x & 31;
    const int waveId = blockIdx.x * 8 + (threadIdx.x >> 5); // 0..4095
    const int tile   = waveId & (NTILES - 1);
    const int slice  = waveId >> 7;

    const int n    = (tile << 4) + (lane & 15);
    const int half = lane >> 4;

    const long k0 = (long)slice * (CH_PER_SLICE * 32) + 16 * half;
    const float*          dmp = dm   + k0 * O_ + n;
    const unsigned short* wp  = weff + (k0 & (E_ - 1)) * O_ + n; // slice never crosses a d boundary
    const unsigned short* ap  = apack + (size_t)slice * CH_PER_SLICE * 512 + lane * 16;

    v8f c = {};
    for (int ch = 0; ch < CH_PER_SLICE; ++ch) {
        BF16Frag a, b;
        a.u = *(const u16x16*)ap;
        #pragma unroll
        for (int j = 0; j < 16; ++j) {
            float         dv = __builtin_nontemporal_load(dmp + (long)j * O_);
            unsigned short w = wp[j * O_];
            b.u[j] = (dv != 0.0f) ? w : (unsigned short)0;
        }
        c = __builtin_amdgcn_wmma_f32_16x16x32_bf16(
                /*neg_a=*/false, a.v, /*neg_b=*/false, b.v,
                /*c_mod=*/(short)0, c, /*reuse_a=*/false, /*reuse_b=*/false);
        ap  += 512;
        dmp += 32L * O_;
        wp  += 32L * O_;
    }
    // C/D layout: lane<16 holds rows M=0..7 (the real batch rows) in elems 0..7
    if (lane < 16) {
        float* pp = partial + (size_t)slice * (B_ * O_) + (tile << 4) + (lane & 15);
        #pragma unroll
        for (int r = 0; r < B_; ++r) pp[r * O_] = c[r];
    }
}

// ---- 4) reduce K-slice partials -> out (fully overwrites d_out) ----
__global__ void reduce_kernel(const float* __restrict__ partial,
                              float* __restrict__ out) {
    int i = blockIdx.x * blockDim.x + threadIdx.x;
    if (i >= B_ * O_) return;
    float s = 0.0f;
    #pragma unroll 8
    for (int sl = 0; sl < KSLICES; ++sl) s += partial[(size_t)sl * (B_ * O_) + i];
    out[i] = s;
}

extern "C" void kernel_launch(void* const* d_in, const int* in_sizes, int n_in,
                              void* d_out, int out_size, void* d_ws, size_t ws_size,
                              hipStream_t stream) {
    const float* W      = (const float*)d_in[0];
    const float* signs  = (const float*)d_in[1];
    const float* Xd     = (const float*)d_in[2];
    const float* Wshort = (const float*)d_in[3];
    const float* dm     = (const float*)d_in[4];
    float* out = (float*)d_out;

    // workspace layout
    unsigned short* weff  = (unsigned short*)d_ws;                       // 8 MB
    unsigned short* apack = weff + (size_t)E_ * O_;                      // 1 MB
    float*          part  = (float*)(apack + (size_t)NCHUNK * 512);      // 2 MB

    weff_kernel <<<(E_ * O_ + 255) / 256, 256, 0, stream>>>(W, signs, weff);
    apack_kernel<<<(KTOT + 255) / 256,     256, 0, stream>>>(Xd, Wshort, apack);
    gemm_kernel <<<(NTILES * KSLICES) / 8, 256, 0, stream>>>(weff, apack, dm, part);
    reduce_kernel<<<(B_ * O_ + 255) / 256, 256, 0, stream>>>(part, out);
}
